// CrissCrossAttention_affine_x_84189948936442
// MI455X (gfx1250) — matrix operand order repaired and checked by reading, not compiled, not run on hardware
//
#include <hip/hip_runtime.h>
#include <cmath>

#define DEVINL __device__ __forceinline__

typedef __attribute__((ext_vector_type(16))) _Float16 v16h;
typedef __attribute__((ext_vector_type(8)))  _Float16 v8h;
typedef __attribute__((ext_vector_type(4)))  _Float16 v4h;
typedef __attribute__((ext_vector_type(8)))  float    v8f;
typedef __attribute__((ext_vector_type(4)))  float    v4f;
typedef __attribute__((ext_vector_type(4)))  unsigned int v4u;
typedef __attribute__((ext_vector_type(4)))  int      v4i;
typedef __attribute__((ext_vector_type(8)))  int      v8i;

static constexpr int Bn  = 4;
static constexpr int Cn  = 256;
static constexpr int Hn  = 128;
static constexpr int Wn  = 128;
static constexpr int RD  = 32;       // C/8
static constexpr int HWn = Hn * Wn;  // 16384
static constexpr int MQ  = RD, MK = RD, MV = Cn;
static constexpr int MT  = MQ + MK + MV; // 320 fused output channels

// Wave-uniform value pinned to an SGPR (threadIdx-derived values otherwise
// live in VGPRs and force exec-mask control flow around uniform branches).
static DEVINL int uwave() {
  return __builtin_amdgcn_readfirstlane((int)(threadIdx.x >> 5));
}

// ---------------------------------------------------------------------------
// WMMA helper: D = A(16x32 f16) * B(32x16 f16) + C(16x16 f32)
// ---------------------------------------------------------------------------
static DEVINL v8f wmma16(v16h a, v16h b, v8f c) {
  return __builtin_amdgcn_wmma_f32_16x16x32_f16(
      /*neg_a=*/false, a, /*neg_b=*/false, b,
      /*c_mod=*/(short)0, c, /*reuse_a=*/false, /*reuse_b=*/false);
}

// Per-lane fragment coordinates used here (wave32; A/B in contiguous k order, not the ISA table order):
//  A 16x32 f16: m = lane&15; elem j -> k = 16*(lane>=16) + j
//  B 32x16 f16: n = lane&15; elem j -> k = 16*(lane>=16) + j
//  C/D 16x16 f32: n = lane&15; elem r -> m = r + 8*(lane>=16)
static DEVINL int a_k(int half, int j) { return (half << 4) + j; }
static DEVINL int b_k(int half, int j) { return (half << 4) + j; }

// Build an A fragment from two contiguous 8-half (16B) chunks
static DEVINL v16h a_frag_contig(const _Float16* row, int k0, int half) {
  v8h a0 = *(const v8h*)(row + k0 + (half << 4));
  v8h a1 = *(const v8h*)(row + k0 + 8 + (half << 4));
  v16h av;
#pragma unroll
  for (int j = 0; j < 8; ++j) { av[j] = a0[j]; av[8 + j] = a1[j]; }
  return av;
}

// Nearest-neighbor rotation gather (matches jnp.round = round-half-even via rintf)
static DEVINL float rot_sample(const float* __restrict__ img, int h, int w,
                               float cs, float sn) {
  const float cx = (Wn - 1) * 0.5f, cy = (Hn - 1) * 0.5f;
  float xo = (float)w - cx, yo = (float)h - cy;
  float sx =  cs * xo + sn * yo + cx;
  float sy = -sn * xo + cs * yo + cy;
  int ix = (int)rintf(sx);
  int iy = (int)rintf(sy);
  if (ix < 0 || ix >= Wn || iy < 0 || iy >= Hn) return 0.0f;
  return img[iy * Wn + ix];
}

// ---------------------------------------------------------------------------
// Pack Wq|Wk|Wv -> f16 [320x256], biases -> f32 [320]
// ---------------------------------------------------------------------------
__global__ void k_pack(const float* __restrict__ Wq, const float* __restrict__ Wk,
                       const float* __restrict__ Wv, const float* __restrict__ bq,
                       const float* __restrict__ bk, const float* __restrict__ bv,
                       _Float16* __restrict__ Wh, float* __restrict__ bias) {
  int gid = blockIdx.x * 256 + threadIdx.x; // < 320*256
  int o = gid >> 8, c = gid & 255;
  float wv;
  if (o < MQ)            wv = Wq[o * Cn + c];
  else if (o < MQ + MK)  wv = Wk[(o - MQ) * Cn + c];
  else                   wv = Wv[(o - MQ - MK) * Cn + c];
  Wh[gid] = (_Float16)wv;
  if (c == 0) {
    float bb;
    if (o < MQ)           bb = bq[o];
    else if (o < MQ + MK) bb = bk[o - MQ];
    else                  bb = bv[o - MQ - MK];
    bias[o] = bb;
  }
}

__global__ void k_tohalf(const v4f* __restrict__ x, v4h* __restrict__ xh) {
  size_t gid = (size_t)blockIdx.x * 256 + threadIdx.x;
  v4f a = x[gid];
  v4h o;
#pragma unroll
  for (int i = 0; i < 4; ++i) o[i] = (_Float16)a[i];
  xh[gid] = o;
}

__global__ void k_rot_to_h(const float* __restrict__ x, _Float16* __restrict__ xh,
                           float cs, float sn) {
  long long gid = (long long)blockIdx.x * 256 + threadIdx.x;
  int w = (int)(gid & (Wn - 1));
  int h = (int)((gid >> 7) & (Hn - 1));
  long long bc = gid >> 14;
  xh[gid] = (_Float16)rot_sample(x + bc * HWn, h, w, cs, sn);
}

// ---------------------------------------------------------------------------
// Fused QKV projection: Y[o,p] = sum_c Wh[o,c]*xh[b,c,p] + bias[o]
// Per block: 64 pixels, all 320 output rows. The 256x64 f16 x-tile (32 KB,
// row stride HW) is staged into LDS by the Tensor Data Mover: wave 0 (scalar
// branch) builds the D# (ISA 8.3/8.4 bitfields), issues tensor_load_to_lds,
// and waits on TENSORcnt before the workgroup barrier.
// ---------------------------------------------------------------------------
__global__ void k_proj(const _Float16* __restrict__ Wh, const float* __restrict__ bias,
                       const _Float16* __restrict__ xh,
                       _Float16* __restrict__ qh, _Float16* __restrict__ kh,
                       _Float16* __restrict__ vh) {
  __shared__ _Float16 tile[Cn * 64]; // [c][n] 32 KB, filled by TDM
  const int b  = blockIdx.y;
  const int p0 = blockIdx.x * 64;
  const _Float16* xb = xh + (size_t)b * Cn * HWn + p0;
  const int wave = uwave();           // SGPR: scalar control flow below
  const int lane = threadIdx.x & 31;

  if (wave == 0) { // scalar s_cbranch: exactly one TDM issue per workgroup
    unsigned long long ga = (unsigned long long)(uintptr_t)xb;
    unsigned int ldsa = (unsigned int)(uintptr_t)(&tile[0]);
    // D# group 0: count=1 | lds_addr | global_addr[56:0] | type=2
    v4u g0;
    g0[0] = 1u;
    g0[1] = ldsa;
    g0[2] = (unsigned int)ga;
    g0[3] = (unsigned int)((ga >> 32) & 0x01FFFFFFull) | (2u << 30);
    // D# group 1: data_size=1 (2B); tensor_dim0=HW, tensor_dim1=C;
    // tile_dim0=64, tile_dim1=C; tensor_dim0_stride=HW
    v8i g1;
    g1[0] = (int)(1u << 16);
    g1[1] = (int)((unsigned)(HWn & 0xFFFF) << 16);
    g1[2] = (int)((unsigned)(HWn >> 16) | ((unsigned)(Cn & 0xFFFF) << 16));
    g1[3] = (int)((unsigned)(Cn >> 16) | (64u << 16));
    g1[4] = (int)Cn;   // tile_dim1=256, tile_dim2=0
    g1[5] = (int)HWn;  // tensor_dim0_stride[31:0]
    g1[6] = 0;
    g1[7] = 0;
    v4i gz; gz[0] = 0; gz[1] = 0; gz[2] = 0; gz[3] = 0;
#if __clang_major__ >= 23
    v8i gz8; gz8[0]=0; gz8[1]=0; gz8[2]=0; gz8[3]=0; gz8[4]=0; gz8[5]=0; gz8[6]=0; gz8[7]=0;
    __builtin_amdgcn_tensor_load_to_lds(g0, g1, gz, gz, gz8, 0);
#else
    __builtin_amdgcn_tensor_load_to_lds(g0, g1, gz, gz, 0);
#endif
    __builtin_amdgcn_s_wait_tensorcnt(0);
  }
  __syncthreads();

  const int half = lane >> 4, l15 = lane & 15;
  for (int t = wave; t < 80; t += 8) {         // scalar loop: 20 M-tiles x 4 N-tiles
    const int m0 = (t >> 2) * 16, n0 = (t & 3) * 16;
    const _Float16* wrow = Wh + (size_t)(m0 + l15) * Cn;
    v8f acc;
#pragma unroll
    for (int r = 0; r < 8; ++r) acc[r] = bias[m0 + r + half * 8];
    for (int k0 = 0; k0 < Cn; k0 += 32) {
      v16h av = a_frag_contig(wrow, k0, half);
      v16h bvv;
#pragma unroll
      for (int j = 0; j < 16; ++j)
        bvv[j] = tile[(k0 + b_k(half, j)) * 64 + n0 + l15];
      acc = wmma16(av, bvv, acc);
    }
    // m0 is an SGPR value -> scalar branch, SGPR base, saddr-form stores
    _Float16* dst; int ob;
    if (m0 < MQ)           { dst = qh + (size_t)b * MQ * HWn; ob = m0; }
    else if (m0 < MQ + MK) { dst = kh + (size_t)b * MK * HWn; ob = m0 - MQ; }
    else                   { dst = vh + (size_t)b * MV * HWn; ob = m0 - MQ - MK; }
    const int p = p0 + n0 + l15;
#pragma unroll
    for (int r = 0; r < 8; ++r)
      dst[(size_t)(ob + r + half * 8) * HWn + p] = (_Float16)acc[r];
  }
}

// ---------------------------------------------------------------------------
// Column energies: eH[b,h,w,g] = sum_c q[b,c,h,w]*k[b,c,g,w]; diag(h==g)=-inf
// ---------------------------------------------------------------------------
__global__ void k_ecol(const _Float16* __restrict__ qh, const _Float16* __restrict__ kh,
                       float* __restrict__ eH) {
  const int b = blockIdx.y, w = blockIdx.x;
  const int wave = uwave(), lane = threadIdx.x & 31;
  const int half = lane >> 4, l15 = lane & 15;
  for (int t = wave; t < 64; t += 8) {
    const int m0 = (t >> 3) * 16, n0 = (t & 7) * 16;
    v16h av, bvv;
#pragma unroll
    for (int j = 0; j < 16; ++j) {
      av[j]  = qh[((size_t)b * RD + a_k(half, j)) * HWn + (m0 + l15) * Wn + w];
      bvv[j] = kh[((size_t)b * RD + b_k(half, j)) * HWn + (n0 + l15) * Wn + w];
    }
    v8f acc;
#pragma unroll
    for (int r = 0; r < 8; ++r) acc[r] = 0.0f;
    acc = wmma16(av, bvv, acc);
#pragma unroll
    for (int r = 0; r < 8; ++r) {
      const int hh = m0 + r + half * 8, gg = n0 + l15;
      float val = (hh == gg) ? -__builtin_inff() : acc[r];
      eH[(((size_t)b * Hn + hh) * Wn + w) * Hn + gg] = val;
    }
  }
}

// Row energies: eW[b,h,w,g] = sum_c q[b,c,h,w]*k[b,c,h,g]
__global__ void k_erow(const _Float16* __restrict__ qh, const _Float16* __restrict__ kh,
                       float* __restrict__ eW) {
  const int b = blockIdx.y, h = blockIdx.x;
  const int wave = uwave(), lane = threadIdx.x & 31;
  const int half = lane >> 4, l15 = lane & 15;
  for (int t = wave; t < 64; t += 8) {
    const int m0 = (t >> 3) * 16, n0 = (t & 7) * 16;
    v16h av, bvv;
#pragma unroll
    for (int j = 0; j < 16; ++j) {
      av[j]  = qh[((size_t)b * RD + a_k(half, j)) * HWn + h * Wn + (m0 + l15)];
      bvv[j] = kh[((size_t)b * RD + b_k(half, j)) * HWn + h * Wn + (n0 + l15)];
    }
    v8f acc;
#pragma unroll
    for (int r = 0; r < 8; ++r) acc[r] = 0.0f;
    acc = wmma16(av, bvv, acc);
#pragma unroll
    for (int r = 0; r < 8; ++r) {
      const int ww = m0 + r + half * 8, gg = n0 + l15;
      eW[(((size_t)b * Hn + h) * Wn + ww) * Wn + gg] = acc[r];
    }
  }
}

// ---------------------------------------------------------------------------
// Joint softmax over [eH(128) | eW(128)] per pixel; one wave per pixel (wave32)
// ---------------------------------------------------------------------------
__global__ void k_softmax(const float* __restrict__ eH, const float* __restrict__ eW,
                          _Float16* __restrict__ aH, _Float16* __restrict__ aW) {
  const int wave = uwave(), lane = threadIdx.x & 31;
  const size_t pix = (size_t)blockIdx.x * 8 + wave; // (b*H+h)*W+w  (scalar)
  const float* ph = eH + pix * Hn;
  const float* pw = eW + pix * Wn;
  float v[8];
  float mx = -3.4e38f;
#pragma unroll
  for (int t = 0; t < 4; ++t) { v[t] = ph[lane + 32 * t]; mx = fmaxf(mx, v[t]); }
#pragma unroll
  for (int t = 0; t < 4; ++t) { v[4 + t] = pw[lane + 32 * t]; mx = fmaxf(mx, v[4 + t]); }
#pragma unroll
  for (int d = 16; d >= 1; d >>= 1) mx = fmaxf(mx, __shfl_xor(mx, d, 32));
  float s = 0.0f;
#pragma unroll
  for (int i = 0; i < 8; ++i) { v[i] = __expf(v[i] - mx); s += v[i]; }
#pragma unroll
  for (int d = 16; d >= 1; d >>= 1) s += __shfl_xor(s, d, 32);
  const float inv = 1.0f / s;
#pragma unroll
  for (int t = 0; t < 4; ++t) aH[pix * Hn + lane + 32 * t] = (_Float16)(v[t] * inv);
#pragma unroll
  for (int t = 0; t < 4; ++t) aW[pix * Wn + lane + 32 * t] = (_Float16)(v[4 + t] * inv);
}

// ---------------------------------------------------------------------------
// Column aggregation: attn[b,c,h,w] = sum_g v[b,c,g,w]*aH[b,h,w,g]
// ---------------------------------------------------------------------------
__global__ void k_aggcol(const _Float16* __restrict__ vh, const _Float16* __restrict__ aH,
                         float* __restrict__ attn) {
  const int b = blockIdx.y, w = blockIdx.x;
  const int wave = uwave(), lane = threadIdx.x & 31;
  const int half = lane >> 4, l15 = lane & 15;
  for (int t = wave; t < 128; t += 8) {        // 16 c-tiles x 8 h-tiles
    const int m0 = (t >> 3) * 16, n0 = (t & 7) * 16;
    const _Float16* arow = aH + (((size_t)b * Hn + n0 + l15) * Wn + w) * Hn;
    v8f acc;
#pragma unroll
    for (int r = 0; r < 8; ++r) acc[r] = 0.0f;
    for (int k0 = 0; k0 < Hn; k0 += 32) {
      v16h av;
#pragma unroll
      for (int j = 0; j < 16; ++j)
        av[j] = vh[((size_t)b * Cn + m0 + l15) * HWn + (k0 + a_k(half, j)) * Wn + w];
      v16h bvv = *(const v16h*)(arow + k0 + (half << 4)); // 32B contiguous
      acc = wmma16(av, bvv, acc);
    }
#pragma unroll
    for (int r = 0; r < 8; ++r) {
      const int cc = m0 + r + half * 8, hh = n0 + l15;
      attn[(((size_t)b * Cn + cc) * Hn + hh) * Wn + w] = acc[r];
    }
  }
}

// Row aggregation: attn[b,c,h,w] += sum_g v[b,c,h,g]*aW[b,h,w,g]
__global__ void k_aggrow(const _Float16* __restrict__ vh, const _Float16* __restrict__ aW,
                         float* __restrict__ attn) {
  const int b = blockIdx.y, h = blockIdx.x;
  const int wave = uwave(), lane = threadIdx.x & 31;
  const int half = lane >> 4, l15 = lane & 15;
  for (int t = wave; t < 128; t += 8) {        // 16 c-tiles x 8 w-tiles
    const int m0 = (t >> 3) * 16, n0 = (t & 7) * 16;
    const _Float16* vrow = vh + ((size_t)b * Cn + m0 + l15) * HWn + h * Wn;
    const _Float16* arow = aW + (((size_t)b * Hn + h) * Wn + (n0 + l15)) * Wn;
    v8f acc;
#pragma unroll
    for (int r = 0; r < 8; ++r) acc[r] = 0.0f;
    for (int k0 = 0; k0 < Wn; k0 += 32) {
      v16h av  = a_frag_contig(vrow, k0, half);           // 2 x 16B contiguous
      v16h bvv = *(const v16h*)(arow + k0 + (half << 4)); // 32B contiguous
      acc = wmma16(av, bvv, acc);
    }
#pragma unroll
    for (int r = 0; r < 8; ++r) {
      const int cc = m0 + r + half * 8, ww = n0 + l15;
      attn[(((size_t)b * Cn + cc) * Hn + h) * Wn + ww] += acc[r];
    }
  }
}

// out = x + g1*attn1 + g2*rot(attn2, +45deg)
__global__ void k_combine(const float* __restrict__ x, const float* __restrict__ a1,
                          const float* __restrict__ a2, const float* __restrict__ g1,
                          const float* __restrict__ g2, float* __restrict__ out,
                          float cs, float sn) {
  long long gid = (long long)blockIdx.x * 256 + threadIdx.x;
  int w = (int)(gid & (Wn - 1));
  int h = (int)((gid >> 7) & (Hn - 1));
  long long bc = gid >> 14;
  float r2 = rot_sample(a2 + bc * HWn, h, w, cs, sn);
  out[gid] = x[gid] + g1[0] * a1[gid] + g2[0] * r2;
}

// ---------------------------------------------------------------------------
extern "C" void kernel_launch(void* const* d_in, const int* in_sizes, int n_in,
                              void* d_out, int out_size, void* d_ws, size_t ws_size,
                              hipStream_t stream) {
  (void)in_sizes; (void)n_in; (void)out_size; (void)ws_size;
  const float* x  = (const float*)d_in[0];
  const float* Wq = (const float*)d_in[1];
  const float* bq = (const float*)d_in[2];
  const float* Wk = (const float*)d_in[3];
  const float* bk = (const float*)d_in[4];
  const float* Wv = (const float*)d_in[5];
  const float* bv = (const float*)d_in[6];
  const float* g1 = (const float*)d_in[7];
  const float* g2 = (const float*)d_in[8];
  float* out = (float*)d_out;

  char* ws = (char*)d_ws;
  size_t off = 0;
  auto take = [&](size_t bytes) -> char* {
    char* p = ws + off;
    off += (bytes + 255) & ~(size_t)255;
    return p;
  };
  _Float16* Wh   = (_Float16*)take((size_t)MT * Cn * 2);
  float*    bias = (float*)   take((size_t)MT * 4);
  _Float16* xh   = (_Float16*)take((size_t)Bn * Cn * HWn * 2);
  _Float16* qh   = (_Float16*)take((size_t)Bn * RD * HWn * 2);
  _Float16* kh   = (_Float16*)take((size_t)Bn * RD * HWn * 2);
  _Float16* vh   = (_Float16*)take((size_t)Bn * Cn * HWn * 2);
  float*    eH   = (float*)   take((size_t)Bn * HWn * Hn * 4);
  float*    eW   = (float*)   take((size_t)Bn * HWn * Wn * 4);
  _Float16* aH   = (_Float16*)take((size_t)Bn * HWn * Hn * 2);
  _Float16* aW   = (_Float16*)take((size_t)Bn * HWn * Wn * 2);
  float*    attn1 = (float*)  take((size_t)Bn * Cn * HWn * 4);
  float*    attn2 = eH; // energies dead after softmax; eH+eW spans 64MB contiguously

  const double PI = 3.14159265358979323846;
  const float csm = (float)cos(-45.0 * PI / 180.0), snm = (float)sin(-45.0 * PI / 180.0);
  const float csp = (float)cos( 45.0 * PI / 180.0), snp = (float)sin( 45.0 * PI / 180.0);

  dim3 blk(256);
  k_pack<<<(MT * Cn) / 256, blk, 0, stream>>>(Wq, Wk, Wv, bq, bk, bv, Wh, bias);

  auto run_attn = [&](float* attn) {
    k_proj   <<<dim3(HWn / 64, Bn), blk, 0, stream>>>(Wh, bias, xh, qh, kh, vh);
    k_ecol   <<<dim3(Wn, Bn),       blk, 0, stream>>>(qh, kh, eH);
    k_erow   <<<dim3(Hn, Bn),       blk, 0, stream>>>(qh, kh, eW);
    k_softmax<<<(Bn * HWn) / 8,     blk, 0, stream>>>(eH, eW, aH, aW);
    k_aggcol <<<dim3(Wn, Bn),       blk, 0, stream>>>(vh, aH, attn);
    k_aggrow <<<dim3(Hn, Bn),       blk, 0, stream>>>(vh, aW, attn);
  };

  // Pass 1: attention on x
  k_tohalf<<<(Bn * Cn * HWn) / (4 * 256), blk, 0, stream>>>((const v4f*)x, (v4h*)xh);
  run_attn(attn1);
  // Pass 2: attention on rot(x, -45)
  k_rot_to_h<<<(Bn * Cn * HWn) / 256, blk, 0, stream>>>(x, xh, csm, snm);
  run_attn(attn2);
  // Final combine with rot(attn2, +45)
  k_combine<<<(Bn * Cn * HWn) / 256, blk, 0, stream>>>(x, attn1, attn2, g1, g2, out, csp, snp);
}
